// LoRAAdapter_19464791785689
// MI455X (gfx1250) — compile-verified, matrix operand order checked by reading, not run on hardware
//
#include <hip/hip_runtime.h>
#include <math.h>
#include <stdint.h>

#define DIM      1024
#define HID      4096
#define MTOK     16384
#define RANK     8
#define SCALINGF 0.125f
#define LDS_STRIDE 40   // 32 bf16 + 8 pad halves; 80B rows keep 16B alignment

typedef __bf16 bf16_t;
typedef __attribute__((ext_vector_type(8)))  bf16_t v8bf;
typedef __attribute__((ext_vector_type(16))) bf16_t v16bf;
typedef __attribute__((ext_vector_type(8)))  float  v8f;

// ---- bf16 <-> f32 helpers (round-to-nearest-even via bit ops) ----
__device__ __forceinline__ bf16_t f2bf(float f) {
    union { float f; unsigned u; } v; v.f = f;
    unsigned r = v.u + 0x7FFFu + ((v.u >> 16) & 1u);
    unsigned short h = (unsigned short)(r >> 16);
    bf16_t b; __builtin_memcpy(&b, &h, 2);
    return b;
}
__device__ __forceinline__ float bf2f(bf16_t b) {
    unsigned short h; __builtin_memcpy(&h, &b, 2);
    union { unsigned u; float f; } v; v.u = ((unsigned)h) << 16;
    return v.f;
}

// ---- one-time f32 -> bf16 tensor conversion (vectorized 8 elems/thread) ----
__global__ __launch_bounds__(256) void cvt_f32_bf16_kernel(const float* __restrict__ src,
                                                           bf16_t* __restrict__ dst,
                                                           int n8) {
    const int i = blockIdx.x * 256 + threadIdx.x;
    if (i >= n8) return;
    const float4* s = (const float4*)src + 2 * (size_t)i;
    const float4 a = s[0];
    const float4 b = s[1];
    v8bf c;
    c[0] = f2bf(a.x); c[1] = f2bf(a.y); c[2] = f2bf(a.z); c[3] = f2bf(a.w);
    c[4] = f2bf(b.x); c[5] = f2bf(b.y); c[6] = f2bf(b.z); c[7] = f2bf(b.w);
    ((v8bf*)dst)[i] = c;
}

// ---- async tile staging: GLOBAL_LOAD_ASYNC_TO_LDS_B128, ASYNCcnt-tracked ----
// Generic pointer to __shared__ carries the wave-relative LDS offset in its
// low 32 bits (aperture addressing). INST_OFFSET applies to BOTH the LDS and
// global address (ISA 15.18.3), so one base pair covers each 16B chunk.

// 128 rows x 32 halves: each of 256 threads copies a half-row (32B).
__device__ __forceinline__ void stage_async_128(const bf16_t* __restrict__ src, int ld,
                                                int row0, int k0, bf16_t* dst, int tid) {
    const int row = tid >> 1;
    const int hk  = (tid & 1) << 4;
    const bf16_t* g = src + (size_t)(row0 + row) * ld + k0 + hk;
    const unsigned loff = (unsigned)(uintptr_t)(dst + row * LDS_STRIDE + hk);
    asm volatile("global_load_async_to_lds_b128 %0, %1, off"
                 :: "v"(loff), "v"(g) : "memory");
    asm volatile("global_load_async_to_lds_b128 %0, %1, off offset:16"
                 :: "v"(loff), "v"(g) : "memory");
}

// 256 rows x 32 halves: each of 256 threads copies one full row (64B).
__device__ __forceinline__ void stage_async_256(const bf16_t* __restrict__ src, int ld,
                                                int row0, int k0, bf16_t* dst, int tid) {
    const bf16_t* g = src + (size_t)(row0 + tid) * ld + k0;
    const unsigned loff = (unsigned)(uintptr_t)(dst + tid * LDS_STRIDE);
    asm volatile("global_load_async_to_lds_b128 %0, %1, off"
                 :: "v"(loff), "v"(g) : "memory");
    asm volatile("global_load_async_to_lds_b128 %0, %1, off offset:16"
                 :: "v"(loff), "v"(g) : "memory");
    asm volatile("global_load_async_to_lds_b128 %0, %1, off offset:32"
                 :: "v"(loff), "v"(g) : "memory");
    asm volatile("global_load_async_to_lds_b128 %0, %1, off offset:48"
                 :: "v"(loff), "v"(g) : "memory");
}

__device__ __forceinline__ void wait_async_all() {
    asm volatile("s_wait_asynccnt 0x0" ::: "memory");
}

// ---- load one 16x32 bf16 WMMA operand fragment from an LDS tile ----
// ISA layout: lanes 0-15 = row, K = {0..7, 16..23}; lanes 16-31: K = {8..15, 24..31}
__device__ __forceinline__ v16bf load_frag(const bf16_t* tile, int row16, int lane) {
    const int row = lane & 15;
    const int kb  = (lane >> 4) << 3;    // 0 or 8
    const bf16_t* p = tile + (row16 + row) * LDS_STRIDE + kb;
    v16bf f;
    ((v8bf*)&f)[0] = *(const v8bf*)(p);        // K = kb .. kb+7
    ((v8bf*)&f)[1] = *(const v8bf*)(p + 16);   // K = kb+16 .. kb+23
    return f;
}

// ---- one K=32 step: 4 A-frags x 4 B-frags -> 16 WMMAs ----
__device__ __forceinline__ void mma_step(const bf16_t* tA, const bf16_t* tB,
                                         int wrow, int wcol, int lane, v8f acc[4][4]) {
    v16bf afrag[4];
    #pragma unroll
    for (int i = 0; i < 4; ++i)
        afrag[i] = load_frag(tA, wrow * 64 + i * 16, lane);
    v16bf bfrag[4];
    #pragma unroll
    for (int j = 0; j < 4; ++j)
        bfrag[j] = load_frag(tB, wcol * 64 + j * 16, lane);
    #pragma unroll
    for (int i = 0; i < 4; ++i)
        #pragma unroll
        for (int j = 0; j < 4; ++j)
            acc[i][j] = __builtin_amdgcn_wmma_f32_16x16x32_bf16(
                false, afrag[i], false, bfrag[j], (short)0, acc[i][j],
                false, false);
}

// ---- tiny rank-8 projection: T[m, r] = sum_k X[m,k] * A[k,r], X in bf16 ----
__global__ __launch_bounds__(256) void lora_proj_kernel(const bf16_t* __restrict__ X,
                                                        const float* __restrict__ A,
                                                        float* __restrict__ T, int K) {
    const int m = blockIdx.x;
    const int tid = threadIdx.x;
    const bf16_t* x = X + (size_t)m * K;
    float acc[RANK] = {};
    for (int k = tid; k < K; k += 256) {
        const float xv = bf2f(x[k]);
        const float* a = A + (size_t)k * RANK;
        #pragma unroll
        for (int r = 0; r < RANK; ++r) acc[r] += xv * a[r];
    }
    #pragma unroll
    for (int r = 0; r < RANK; ++r)
        #pragma unroll
        for (int off = 16; off > 0; off >>= 1)
            acc[r] += __shfl_xor(acc[r], off, 32);
    __shared__ float sred[8][RANK];
    const int wid = tid >> 5, lane = tid & 31;
    if (lane == 0) {
        #pragma unroll
        for (int r = 0; r < RANK; ++r) sred[wid][r] = acc[r];
    }
    __syncthreads();
    if (tid < RANK) {
        float s = 0.f;
        #pragma unroll
        for (int w = 0; w < 8; ++w) s += sred[w][tid];
        T[(size_t)m * RANK + tid] = s;
    }
}

// ---- main fused GEMM: out = act(A @ W^T + b + s*T@Bl [+ resid]) ----
// Block: 256 thr (8 waves), 128x256 C tile. Wave = 64x64 = 4x4 WMMA frags.
// Both operands bf16, async-staged to LDS, double-buffered, pipelined.
template <bool IS_DOWN>
__global__ __launch_bounds__(256) void lora_gemm_kernel(
    const bf16_t* __restrict__ Aop, const bf16_t* __restrict__ W,
    const float* __restrict__ bias, const float* __restrict__ T,
    const float* __restrict__ Bl, const float* __restrict__ resid,
    void* __restrict__ outp, int N, int K) {
    __shared__ bf16_t sA[2][128 * LDS_STRIDE];   // 20 KB
    __shared__ bf16_t sB[2][256 * LDS_STRIDE];   // 40 KB

    const int tid  = threadIdx.x;
    const int lane = tid & 31;
    const int wid  = tid >> 5;
    const int wrow = wid & 1;    // 2 wave-rows of 64
    const int wcol = wid >> 1;   // 4 wave-cols of 64
    const int n0 = blockIdx.x * 256;
    const int m0 = blockIdx.y * 128;

    v8f acc[4][4] = {};

    // prologue stage
    stage_async_128(Aop, K, m0, 0, sA[0], tid);
    stage_async_256(W,   K, n0, 0, sB[0], tid);
    wait_async_all();
    __syncthreads();

    const int nk = K >> 5;
    for (int ks = 0; ks < nk - 1; ++ks) {
        const int cur = ks & 1;
        const int k1 = (ks + 1) << 5;
        stage_async_128(Aop, K, m0, k1, sA[cur ^ 1], tid);   // prefetch next slab
        stage_async_256(W,   K, n0, k1, sB[cur ^ 1], tid);
        mma_step(sA[cur], sB[cur], wrow, wcol, lane, acc);   // compute current
        wait_async_all();
        __syncthreads();
    }
    mma_step(sA[(nk - 1) & 1], sB[(nk - 1) & 1], wrow, wcol, lane, acc);  // drain

    // epilogue: bias + rank-8 LoRA correction (+ gelu / + residual)
    const int ncol = n0 + wcol * 64;
    #pragma unroll
    for (int i = 0; i < 4; ++i) {
        const int mbase = m0 + wrow * 64 + i * 16 + ((lane >> 4) << 3);
        #pragma unroll
        for (int v = 0; v < 8; ++v) {
            const int mm = mbase + v;
            const float* t = T + (size_t)mm * RANK;
            float tl[RANK];
            #pragma unroll
            for (int r = 0; r < RANK; ++r) tl[r] = t[r];
            #pragma unroll
            for (int j = 0; j < 4; ++j) {
                const int nn = ncol + j * 16 + (lane & 15);
                float val = acc[i][j][v] + bias[nn];
                float l = 0.f;
                #pragma unroll
                for (int r = 0; r < RANK; ++r) l += tl[r] * Bl[r * N + nn];
                val += SCALINGF * l;
                const size_t idx = (size_t)mm * N + nn;
                if (IS_DOWN) {
                    val = 0.5f * val * (1.0f + erff(val * 0.70710678118654752f));
                    ((bf16_t*)outp)[idx] = f2bf(val);
                } else {
                    val += resid[idx];
                    ((float*)outp)[idx] = val;
                }
            }
        }
    }
}

extern "C" void kernel_launch(void* const* d_in, const int* in_sizes, int n_in,
                              void* d_out, int out_size, void* d_ws, size_t ws_size,
                              hipStream_t stream) {
    const float* x      = (const float*)d_in[0];
    const float* W_down = (const float*)d_in[1];
    const float* b_down = (const float*)d_in[2];
    const float* A_down = (const float*)d_in[3];
    const float* B_down = (const float*)d_in[4];
    const float* W_up   = (const float*)d_in[5];
    const float* b_up   = (const float*)d_in[6];
    const float* A_up   = (const float*)d_in[7];
    const float* B_up   = (const float*)d_in[8];
    float* out = (float*)d_out;

    // workspace layout
    char* ws = (char*)d_ws;
    size_t off = 0;
    float*  T1  = (float*)(ws + off);  off += (size_t)MTOK * RANK * 4;      // 512 KB
    float*  T2  = (float*)(ws + off);  off += (size_t)MTOK * RANK * 4;      // 512 KB
    bf16_t* xb  = (bf16_t*)(ws + off); off += (size_t)MTOK * DIM * 2;       // 32 MB
    bf16_t* Wdb = (bf16_t*)(ws + off); off += (size_t)HID * DIM * 2;        // 8 MB
    bf16_t* Wub = (bf16_t*)(ws + off); off += (size_t)DIM * HID * 2;        // 8 MB
    bf16_t* h   = (bf16_t*)(ws + off);                                      // 128 MB

    // 0) one-time f32 -> bf16 conversions (bandwidth-bound, ~5us total)
    {
        const int nx = MTOK * DIM / 8, nw = HID * DIM / 8;
        cvt_f32_bf16_kernel<<<(nx + 255) / 256, 256, 0, stream>>>(x, xb, nx);
        cvt_f32_bf16_kernel<<<(nw + 255) / 256, 256, 0, stream>>>(W_down, Wdb, nw);
        cvt_f32_bf16_kernel<<<(nw + 255) / 256, 256, 0, stream>>>(W_up, Wub, nw);
    }
    // 1) T1 = x @ A_down
    lora_proj_kernel<<<MTOK, 256, 0, stream>>>(xb, A_down, T1, DIM);
    // 2) h = gelu(x @ W_down^T + b_down + s * T1 @ B_down), stored bf16
    lora_gemm_kernel<true><<<dim3(HID / 256, MTOK / 128), 256, 0, stream>>>(
        xb, Wdb, b_down, T1, B_down, nullptr, (void*)h, HID, DIM);
    // 3) T2 = h @ A_up
    lora_proj_kernel<<<MTOK, 256, 0, stream>>>(h, A_up, T2, HID);
    // 4) out = h @ W_up^T + b_up + s * T2 @ B_up + x
    lora_gemm_kernel<false><<<dim3(DIM / 256, MTOK / 128), 256, 0, stream>>>(
        h, Wub, b_up, T2, B_up, x, (void*)out, DIM, HID);
}
